// MHLA2_73572789781029
// MI455X (gfx1250) — matrix operand order repaired and checked by reading, compiled, not run on hardware
//
#include <hip/hip_runtime.h>

#define NHEADS 8
#define DIN    512
#define DQ     64
#define BATCH  4
#define NSEQ   8192
#define HDIM   512                      // NHEADS*DQ
#define KEXPSCALE 0.51008893f           // log2(e) / 64^0.25

typedef __attribute__((ext_vector_type(16))) _Float16     v16h;
typedef __attribute__((ext_vector_type(8)))  float        v8f;
typedef __attribute__((ext_vector_type(8)))  unsigned int v8u;

// ---------------- WMMA fragment helpers (ISA 7.12.2 layouts) ----------------
// A fragment (f16 16x32): lanes 0-15 row M=lane, dwords {0..3, 8..11} of the
// 16-dword K row; lanes 16-31 row M=lane-16, dwords {4..7, 12..15}.
__device__ __forceinline__ v16h ldsA(const unsigned int* base, int row,
                                     int stride_dw, int kd, int lane) {
  const unsigned int* r = base + (size_t)(row + (lane & 15)) * stride_dw + kd +
                          ((lane & 16) ? 4 : 0);
  v8u d;
#pragma unroll
  for (int j = 0; j < 4; ++j) { d[j] = r[j]; d[4 + j] = r[8 + j]; }
  union { v8u u; v16h h; } cvt; cvt.u = d; return cvt.h;
}
// B fragment (f16 32x16): lane holds column N=lane&15; lanes 0-15 dwords
// {0..3, 8..11} of the column's k-array, lanes 16-31 dwords {4..7, 12..15}.
// Requires the B tile staged transposed in LDS: bt[col][k].
__device__ __forceinline__ v16h ldsB(const unsigned int* base, int col,
                                     int stride_dw, int kd, int lane) {
  const unsigned int* r = base + (size_t)(col + (lane & 15)) * stride_dw + kd +
                          ((lane & 16) ? 4 : 0);
  v8u d;
#pragma unroll
  for (int j = 0; j < 4; ++j) { d[j] = r[j]; d[4 + j] = r[8 + j]; }
  union { v8u u; v16h h; } cvt; cvt.u = d; return cvt.h;
}
__device__ __forceinline__ v8f wmma_f16(v16h a, v16h b, v8f c) {
  return __builtin_amdgcn_wmma_f32_16x16x32_f16(false, a, false, b, (short)0, c,
                                                false, false);
}

// ---------------- weight packing ----------------
// W_[h][k][q] (f32) -> Wp[k][h*64+q] (f16): B operand of the projection GEMM.
__global__ __launch_bounds__(256) void pack_qkv(const float* __restrict__ w,
                                                _Float16* __restrict__ dst) {
  int idx = blockIdx.x * 256 + threadIdx.x;           // 8*512*64 = 262144
  int h = idx >> 15, r = idx & 32767, k = r >> 6, q = r & 63;
  dst[(size_t)k * HDIM + h * 64 + q] = (_Float16)w[idx];
}
// W_O[c][i] (f32) -> WoT[i][c] (f16): B operand of the output GEMM.
__global__ __launch_bounds__(256) void pack_wo(const float* __restrict__ w,
                                               _Float16* __restrict__ dst) {
  int idx = blockIdx.x * 256 + threadIdx.x;           // 512*512
  int c = idx >> 9, i = idx & 511;
  dst[(size_t)i * HDIM + c] = (_Float16)w[idx];
}

// ---------------- QKV projection: Out[b][n][h*64+q] = sum_d X[b][d][n] W[d][hq]
// M=8192, N=512, K=512 per batch. Workgroup tile 128m x 128c, 8 waves (4x2),
// each wave 32m x 64c = 8 accum fragments. fp32->f16 transpose fused in A-stage.
__global__ __launch_bounds__(256) void proj_gemm(const float* __restrict__ X,
                                                 const _Float16* __restrict__ Wp,
                                                 _Float16* __restrict__ Out) {
  int b  = blockIdx.z;
  int mb = blockIdx.x * 128;
  int cb = blockIdx.y * 128;
  const float* Xb = X + (size_t)b * DIN * NSEQ;
  _Float16*    Ob = Out + (size_t)b * NSEQ * HDIM;
  __shared__ unsigned int lds[2176 * 2];              // am[128][17] | wt[128][17]
  unsigned int* am = lds;
  unsigned int* wt = lds + 2176;
  _Float16* amh = (_Float16*)am;
  _Float16* wth = (_Float16*)wt;
  int t = threadIdx.x, lane = t & 31, w = t >> 5;
  int wm = (w & 3) * 32;                              // wave m offset (2 tiles)
  int wc = (w >> 2) * 64;                             // wave c offset (4 tiles)
  v8f acc[2][4];
  for (int i = 0; i < 2; ++i)
    for (int j = 0; j < 4; ++j)
      for (int e = 0; e < 8; ++e) acc[i][j][e] = 0.0f;

  for (int kb = 0; kb < DIN; kb += 32) {
    __syncthreads();
    // Stage A transposed+converted: am[n][k] = (f16) X[kb+k][mb+n]
#pragma unroll
    for (int i = 0; i < 16; ++i) {
      int e = t + 256 * i, k = e >> 7, n = e & 127;
      amh[n * 34 + k] = (_Float16)Xb[(size_t)(kb + k) * NSEQ + mb + n];
    }
    // Stage B transposed: wt[c][k] = Wp[kb+k][cb+c]
#pragma unroll
    for (int i = 0; i < 16; ++i) {
      int e = t + 256 * i, k = e >> 7, c = e & 127;
      wth[c * 34 + k] = Wp[(size_t)(kb + k) * HDIM + cb + c];
    }
    if (kb + 32 < DIN)
      __builtin_prefetch((const void*)(Xb + (size_t)(kb + 32) * NSEQ + mb + (lane << 2)), 0, 0);
    __syncthreads();
    v16h a0 = ldsA(am, wm, 17, 0, lane);
    v16h a1 = ldsA(am, wm + 16, 17, 0, lane);
#pragma unroll
    for (int ct = 0; ct < 4; ++ct) {
      v16h bf = ldsB(wt, wc + ct * 16, 17, 0, lane);
      acc[0][ct] = wmma_f16(a0, bf, acc[0][ct]);
      acc[1][ct] = wmma_f16(a1, bf, acc[1][ct]);
    }
  }
  // Epilogue: f32 accum -> f16 row-major [m][512]
#pragma unroll
  for (int mt = 0; mt < 2; ++mt)
#pragma unroll
    for (int ct = 0; ct < 4; ++ct)
#pragma unroll
      for (int j = 0; j < 8; ++j) {
        int c = cb + wc + ct * 16 + (lane & 15);
        int m = mb + wm + mt * 16 + j + ((lane >> 4) & 1) * 8;
        Ob[(size_t)m * HDIM + c] = (_Float16)acc[mt][ct][j];
      }
}

// ---------------- column max over n (softmax pass 1), column-owning blocks ----
__global__ __launch_bounds__(256) void colmax(const _Float16* __restrict__ In,
                                              float* __restrict__ Max) {
  int b = blockIdx.y, cb = blockIdx.x * 32;
  const _Float16* P = In + (size_t)b * NSEQ * HDIM;
  int t = threadIdx.x, c = cb + (t & 31), g = t >> 5;
  float mx = -3.0e38f;
  for (int n = g; n < NSEQ; n += 8)
    mx = fmaxf(mx, (float)P[(size_t)n * HDIM + c]);
  __shared__ float red[8][32];
  red[g][t & 31] = mx;
  __syncthreads();
  if (t < 32) {
    float m = red[0][t];
    for (int r = 1; r < 8; ++r) m = fmaxf(m, red[r][t]);
    Max[b * HDIM + cb + t] = m;
  }
}

// ---------------- exp (Q, in place) + deterministic per-block partial sums ----
__global__ __launch_bounds__(256) void expq_kernel(_Float16* __restrict__ P,
                                                   const float* __restrict__ Max,
                                                   float* __restrict__ SumP) {
  int b = blockIdx.y, nb = blockIdx.x;                // 64-row tile
  _Float16* Q = P + (size_t)b * NSEQ * HDIM + (size_t)nb * 64 * HDIM;
  int t = threadIdx.x;
  float m0 = Max[b * HDIM + t], m1 = Max[b * HDIM + t + 256];
  float s0 = 0.0f, s1 = 0.0f;
  for (int i = 0; i < 128; ++i) {
    int e = t + 256 * i, c = e & 511, n = e >> 9;
    float f  = (float)Q[(size_t)n * HDIM + c];
    float mm = (c == t) ? m0 : m1;
    float ex = exp2f((f - mm) * KEXPSCALE);
    Q[(size_t)n * HDIM + c] = (_Float16)ex;
    if (c == t) s0 += ex; else s1 += ex;
  }
  SumP[((size_t)b * 128 + nb) * 512 + t]       = s0;
  SumP[((size_t)b * 128 + nb) * 512 + t + 256] = s1;
}

// ---------------- exp (K) + transpose to EkT[b][c][n] via LDS ----------------
__global__ __launch_bounds__(256) void expkT_kernel(const _Float16* __restrict__ Kh,
                                                    const float* __restrict__ Max,
                                                    float* __restrict__ SumP,
                                                    _Float16* __restrict__ EkT) {
  int b = blockIdx.y, nb = blockIdx.x;                // 32-row tile
  const _Float16* Kb = Kh + (size_t)b * NSEQ * HDIM + (size_t)nb * 32 * HDIM;
  __shared__ _Float16 et[512 * 34];                   // [c][32 n + pad]
  int t = threadIdx.x;
  float m0 = Max[b * HDIM + t], m1 = Max[b * HDIM + t + 256];
  float s0 = 0.0f, s1 = 0.0f;
  for (int i = 0; i < 64; ++i) {
    int e = t + 256 * i, c = e & 511, n = e >> 9;
    float f  = (float)Kb[(size_t)n * HDIM + c];
    float mm = (c == t) ? m0 : m1;
    float ex = exp2f((f - mm) * KEXPSCALE);
    et[c * 34 + n] = (_Float16)ex;
    if (c == t) s0 += ex; else s1 += ex;
  }
  SumP[((size_t)b * 256 + nb) * 512 + t]       = s0;
  SumP[((size_t)b * 256 + nb) * 512 + t + 256] = s1;
  __syncthreads();
  _Float16* Eb = EkT + (size_t)b * HDIM * NSEQ + (size_t)nb * 32;
  for (int i = 0; i < 64; ++i) {
    int e = t + 256 * i, c = e >> 5, n = e & 31;
    Eb[(size_t)c * NSEQ + n] = et[c * 34 + n];
  }
}

// ---------------- deterministic partial-sum reduction ----------------
__global__ __launch_bounds__(256) void reduce_sum(const float* __restrict__ P,
                                                  float* __restrict__ S, int nblk) {
  int idx = blockIdx.x * 256 + threadIdx.x;           // BATCH*512
  int b = idx >> 9, c = idx & 511;
  float s = 0.0f;
  for (int j = 0; j < nblk; ++j) s += P[((size_t)b * nblk + j) * 512 + c];
  S[idx] = s;
}

// ---------------- A_raw[q][v] = sum_n Ek[n][q] V[n][v], K-split over grid.y ---
// Per block: 64x64 output, K range 1024. Partials to AP[ks][bh][4096] (no atomics).
__global__ __launch_bounds__(256) void a_gemm(const _Float16* __restrict__ EkT,
                                              const _Float16* __restrict__ Vh,
                                              float* __restrict__ AP) {
  int bh = blockIdx.x, ks = blockIdx.y;
  int b = bh >> 3, hc = (bh & 7) * 64;
  int kb0 = ks * 1024;
  const _Float16* E = EkT + (size_t)b * HDIM * NSEQ + (size_t)hc * NSEQ;
  const _Float16* V = Vh + (size_t)b * NSEQ * HDIM;
  __shared__ unsigned int lds2[1088 * 2];             // ek[64][17] | vt[64][17]
  unsigned int* ek = lds2;
  unsigned int* vt = lds2 + 1088;
  _Float16* ekh = (_Float16*)ek;
  _Float16* vth = (_Float16*)vt;
  int t = threadIdx.x, lane = t & 31, w = t >> 5;
  int qt  = (w & 3) * 16;
  int vtb = (w >> 2) * 32;                            // 2 v-tiles per wave
  v8f acc[2];
  for (int i = 0; i < 2; ++i)
    for (int j = 0; j < 8; ++j) acc[i][j] = 0.0f;

  for (int kk = 0; kk < 1024; kk += 32) {
    int kb = kb0 + kk;
    __syncthreads();
    {   // ek[q][k] = E[q][kb+k]  (rows contiguous along n)
      int q = t >> 2, i8 = (t & 3) * 8;
#pragma unroll
      for (int j = 0; j < 8; ++j)
        ekh[q * 34 + i8 + j] = E[(size_t)q * NSEQ + kb + i8 + j];
    }
    {   // vt[v][k] = V[kb+k][hc+v]  (transposed stage for B fragments)
      int k = t >> 3, v8 = (t & 7) * 8;
#pragma unroll
      for (int j = 0; j < 8; ++j)
        vth[(v8 + j) * 34 + k] = V[(size_t)(kb + k) * HDIM + hc + v8 + j];
    }
    __syncthreads();
    v16h a = ldsA(ek, qt, 17, 0, lane);
#pragma unroll
    for (int vv = 0; vv < 2; ++vv) {
      v16h bf = ldsB(vt, vtb + vv * 16, 17, 0, lane);
      acc[vv] = wmma_f16(a, bf, acc[vv]);
    }
  }
  float* A = AP + ((size_t)ks * 32 + bh) * 4096;
#pragma unroll
  for (int vv = 0; vv < 2; ++vv)
#pragma unroll
    for (int j = 0; j < 8; ++j) {
      int v = vtb + vv * 16 + (lane & 15);
      int q = qt + j + ((lane >> 4) & 1) * 8;
      A[q * 64 + v] = acc[vv][j];
    }
}

// ---------------- A'' = sum_ks AP / (S_k[q]*S_q[q]); stored transposed [v][q] --
__global__ __launch_bounds__(256) void a_fin(const float* __restrict__ AP,
                                             const float* __restrict__ Sq,
                                             const float* __restrict__ Sk,
                                             _Float16* __restrict__ A2T) {
  int bh = blockIdx.x, b = bh >> 3, hc = (bh & 7) * 64;
  _Float16* D = A2T + (size_t)bh * 4096;
  for (int i = threadIdx.x; i < 4096; i += 256) {
    int q = i >> 6, v = i & 63;
    float a = 0.0f;
    for (int ks = 0; ks < 8; ++ks) a += AP[((size_t)ks * 32 + bh) * 4096 + i];
    float s = 1.0f / (Sk[b * HDIM + hc + q] * Sq[b * HDIM + hc + q]);
    D[v * 64 + q] = (_Float16)(a * s);
  }
}

// ---------------- Bt[b][h][n][v] = sum_q Eq[n][hc+q] A''[q][v] ----------------
__global__ __launch_bounds__(256) void bt_gemm(const _Float16* __restrict__ Eq,
                                               const _Float16* __restrict__ A2T,
                                               _Float16* __restrict__ Bt) {
  int bh = blockIdx.y, b = bh >> 3, hc = (bh & 7) * 64;
  int mb = blockIdx.x * 128;
  const _Float16* Q  = Eq + (size_t)b * NSEQ * HDIM + (size_t)mb * HDIM + hc;
  const _Float16* AT = A2T + (size_t)bh * 4096;       // already [v][q]
  __shared__ unsigned int lds3[128 * 33 + 64 * 33];   // am[128][33] | vt[64][33]
  unsigned int* am = lds3;
  unsigned int* vt = lds3 + 128 * 33;
  _Float16* amh = (_Float16*)am;
  _Float16* vth = (_Float16*)vt;
  int t = threadIdx.x, lane = t & 31, w = t >> 5;
  int wm = (w & 3) * 32, wc = (w >> 2) * 32;
  for (int i = 0; i < 32; ++i) {                      // am[n][k=q], K=64 staged once
    int e = t + 256 * i, n = e >> 6, k = e & 63;
    amh[n * 66 + k] = Q[(size_t)n * HDIM + k];
  }
  for (int i = 0; i < 16; ++i) {                      // vt[v][q] direct copy
    int e = t + 256 * i;
    vth[(e >> 6) * 66 + (e & 63)] = AT[e];
  }
  __syncthreads();
  v8f acc[2][2];
  for (int i = 0; i < 2; ++i)
    for (int j = 0; j < 2; ++j)
      for (int e = 0; e < 8; ++e) acc[i][j][e] = 0.0f;
#pragma unroll
  for (int kc = 0; kc < 2; ++kc) {
    int kd = kc * 16;
    v16h a0 = ldsA(am, wm, 33, kd, lane);
    v16h a1 = ldsA(am, wm + 16, 33, kd, lane);
#pragma unroll
    for (int ct = 0; ct < 2; ++ct) {
      v16h bf = ldsB(vt, wc + ct * 16, 33, kd, lane);
      acc[0][ct] = wmma_f16(a0, bf, acc[0][ct]);
      acc[1][ct] = wmma_f16(a1, bf, acc[1][ct]);
    }
  }
  _Float16* O = Bt + (size_t)bh * NSEQ * 64 + (size_t)mb * 64;
#pragma unroll
  for (int mt = 0; mt < 2; ++mt)
#pragma unroll
    for (int ct = 0; ct < 2; ++ct)
#pragma unroll
      for (int j = 0; j < 8; ++j) {
        int v = wc + ct * 16 + (lane & 15);
        int n = wm + mt * 16 + j + ((lane >> 4) & 1) * 8;
        O[(size_t)n * 64 + v] = (_Float16)acc[mt][ct][j];
      }
}

// ---------------- out[b][c][r] = sum_i BtFlat[b][r][i] W_O[c][i] -------------
// BtFlat viewed [8192][512] IS the reference's Bv (torch .view identity).
__global__ __launch_bounds__(256) void out_gemm(const _Float16* __restrict__ Bt,
                                                const _Float16* __restrict__ WoT,
                                                float* __restrict__ Outp) {
  int b = blockIdx.z, mb = blockIdx.x * 128, cb = blockIdx.y * 128;
  const _Float16* A = Bt + (size_t)b * NSEQ * HDIM;
  __shared__ unsigned int lds4[2176 * 2 + 64 * 129];  // am | wt | outs[64][129]f32
  unsigned int* am = lds4;
  unsigned int* wt = lds4 + 2176;
  float* outs = (float*)(lds4 + 4352);
  _Float16* amh = (_Float16*)am;
  _Float16* wth = (_Float16*)wt;
  int t = threadIdx.x, lane = t & 31, w = t >> 5;
  int wm = (w & 3) * 32, wc = (w >> 2) * 64;
  v8f acc[2][4];
  for (int i = 0; i < 2; ++i)
    for (int j = 0; j < 4; ++j)
      for (int e = 0; e < 8; ++e) acc[i][j][e] = 0.0f;

  for (int kb = 0; kb < HDIM; kb += 32) {
    __syncthreads();
#pragma unroll
    for (int i = 0; i < 16; ++i) {                    // am[m][k] (row-major f16 src)
      int e = t + 256 * i, m = e >> 5, k = e & 31;
      amh[m * 34 + k] = A[(size_t)(mb + m) * HDIM + kb + k];
    }
#pragma unroll
    for (int i = 0; i < 16; ++i) {                    // wt[c][k] = WoT[kb+k][cb+c]
      int e = t + 256 * i, k = e >> 7, c = e & 127;
      wth[c * 34 + k] = WoT[(size_t)(kb + k) * HDIM + cb + c];
    }
    if (kb + 32 < HDIM)
      __builtin_prefetch((const void*)(A + (size_t)(mb + (t >> 1)) * HDIM + kb + 32), 0, 0);
    __syncthreads();
    v16h a0 = ldsA(am, wm, 17, 0, lane);
    v16h a1 = ldsA(am, wm + 16, 17, 0, lane);
#pragma unroll
    for (int ct = 0; ct < 4; ++ct) {
      v16h bf = ldsB(wt, wc + ct * 16, 17, 0, lane);
      acc[0][ct] = wmma_f16(a0, bf, acc[0][ct]);
      acc[1][ct] = wmma_f16(a1, bf, acc[1][ct]);
    }
  }
  // Transposed f32 store via LDS, 64 columns per pass -> coalesced along m.
  float* Ob = Outp + (size_t)b * HDIM * NSEQ;
  for (int p = 0; p < 2; ++p) {
    __syncthreads();
    if ((w >> 2) == p) {
#pragma unroll
      for (int mt = 0; mt < 2; ++mt)
#pragma unroll
        for (int ct = 0; ct < 4; ++ct)
#pragma unroll
          for (int j = 0; j < 8; ++j) {
            int cl = ct * 16 + (lane & 15);           // local c within 64-wide pass
            int ml = wm + mt * 16 + j + ((lane >> 4) & 1) * 8;
            outs[cl * 129 + ml] = acc[mt][ct][j];
          }
    }
    __syncthreads();
    for (int i = 0; i < 32; ++i) {                    // 64*128 floats
      int e = t + 256 * i, c = e >> 7, m = e & 127;
      Ob[(size_t)(cb + p * 64 + c) * NSEQ + mb + m] = outs[c * 129 + m];
    }
  }
}

// ---------------- host launcher ----------------
extern "C" void kernel_launch(void* const* d_in, const int* in_sizes, int n_in,
                              void* d_out, int out_size, void* d_ws, size_t ws_size,
                              hipStream_t stream) {
  (void)in_sizes; (void)n_in; (void)out_size; (void)ws_size;
  const float* x_q = (const float*)d_in[0];
  const float* x_k = (const float*)d_in[1];
  const float* x_v = (const float*)d_in[2];
  const float* W_Q = (const float*)d_in[3];
  const float* W_K = (const float*)d_in[4];
  const float* W_V = (const float*)d_in[5];
  const float* W_O = (const float*)d_in[6];

  char* ws = (char*)d_ws;
  size_t off = 0;
  auto carve = [&](size_t bytes) -> void* {
    void* p = ws + off;
    off += (bytes + 255) & ~(size_t)255;
    return p;
  };
  const size_t big = (size_t)BATCH * NSEQ * HDIM * sizeof(_Float16);   // 32 MiB
  _Float16* Qh   = (_Float16*)carve(big);              // becomes Eq in place
  _Float16* Kh   = (_Float16*)carve(big);
  _Float16* Vh   = (_Float16*)carve(big);
  _Float16* EkT  = (_Float16*)carve(big);              // [b][c][n]
  _Float16* Bt   = (_Float16*)carve(big);              // [b][h][n][64] == Bv flat
  _Float16* Wq   = (_Float16*)carve((size_t)512 * 512 * 2);
  _Float16* Wk   = (_Float16*)carve((size_t)512 * 512 * 2);
  _Float16* Wv   = (_Float16*)carve((size_t)512 * 512 * 2);
  _Float16* WoT  = (_Float16*)carve((size_t)512 * 512 * 2);
  float*    Maxq = (float*)carve((size_t)BATCH * HDIM * 4);
  float*    Maxk = (float*)carve((size_t)BATCH * HDIM * 4);
  float*    Sq   = (float*)carve((size_t)BATCH * HDIM * 4);
  float*    Sk   = (float*)carve((size_t)BATCH * HDIM * 4);
  float*    SqP  = (float*)carve((size_t)BATCH * 128 * 512 * 4);
  float*    SkP  = (float*)carve((size_t)BATCH * 256 * 512 * 4);
  float*    AP   = (float*)carve((size_t)8 * 32 * 4096 * 4);
  _Float16* A2T  = (_Float16*)carve((size_t)32 * 4096 * 2);

  pack_qkv<<<1024, 256, 0, stream>>>(W_Q, Wq);
  pack_qkv<<<1024, 256, 0, stream>>>(W_K, Wk);
  pack_qkv<<<1024, 256, 0, stream>>>(W_V, Wv);
  pack_wo <<<1024, 256, 0, stream>>>(W_O, WoT);

  dim3 gproj(NSEQ / 128, HDIM / 128, BATCH);
  proj_gemm<<<gproj, 256, 0, stream>>>(x_q, Wq, Qh);
  proj_gemm<<<gproj, 256, 0, stream>>>(x_k, Wk, Kh);
  proj_gemm<<<gproj, 256, 0, stream>>>(x_v, Wv, Vh);

  colmax<<<dim3(16, BATCH), 256, 0, stream>>>(Qh, Maxq);
  colmax<<<dim3(16, BATCH), 256, 0, stream>>>(Kh, Maxk);

  expq_kernel <<<dim3(128, BATCH), 256, 0, stream>>>(Qh, Maxq, SqP);
  reduce_sum  <<<(BATCH * HDIM) / 256, 256, 0, stream>>>(SqP, Sq, 128);
  expkT_kernel<<<dim3(256, BATCH), 256, 0, stream>>>(Kh, Maxk, SkP, EkT);
  reduce_sum  <<<(BATCH * HDIM) / 256, 256, 0, stream>>>(SkP, Sk, 256);

  a_gemm<<<dim3(32, 8), 256, 0, stream>>>(EkT, Vh, AP);
  a_fin <<<32, 256, 0, stream>>>(AP, Sq, Sk, A2T);

  bt_gemm<<<dim3(NSEQ / 128, 32), 256, 0, stream>>>(Qh, A2T, Bt);
  out_gemm<<<dim3(NSEQ / 128, HDIM / 128, BATCH), 256, 0, stream>>>(Bt, WoT,
                                                                   (float*)d_out);
}
